// FocusModel_63367947485662
// MI455X (gfx1250) — matrix operand order repaired and checked by reading, not compiled, not run on hardware
//
#include <hip/hip_runtime.h>

// ---------------------------------------------------------------------------
// FocusModel for MI455X (gfx1250): bidir-LSTM encoder x2 + decoder LSTM.
// All matmuls via v_wmma_f32_16x16x32_f16 (f16 operands, f32 accum).
// Weights converted to f16 once (~31MB -> L2-resident across the recurrence).
// Each wave owns a 32(batch) x 16(hidden) tile: 2 row-tiles x 4 gate-tiles,
// so every B (weight) fragment feeds 2 WMMAs (40% less L2 traffic than 1-rt).
// Encoder fwd+bwd directions fused into one launch per timestep (grid.y=dir).
// ---------------------------------------------------------------------------

#define L_SEQ 160
#define N_B   128
#define E_DIM 512
#define HE_D  512
#define HD_D  1024
#define T_TAG 128

typedef _Float16 f16;
typedef _Float16 h8   __attribute__((ext_vector_type(8)));
typedef _Float16 v16h __attribute__((ext_vector_type(16)));
typedef float    v8f  __attribute__((ext_vector_type(8)));

// A operand: 16x32 tile of row-major (M x K). Lane m=lane&15,
// frag[0..7] = A[m][k0 + (lane>>4)*8 .. +7], frag[8..15] = same +16.
__device__ __forceinline__ v16h load_a_frag(const f16* __restrict__ tile,
                                            int ld, int k0, int lane) {
  const int m  = lane & 15;
  const int hl = lane >> 4;
  const f16* p = tile + (size_t)m * ld + k0 + hl * 8;
  h8 lo = *(const h8*)(p);
  h8 hi = *(const h8*)(p + 16);
  v16h f;
#pragma unroll
  for (int i = 0; i < 8; ++i) { f[i] = lo[i]; f[i + 8] = hi[i]; }
  return f;
}

// B operand: 32x16 tile of B, loaded from B^T row-major (N x K) — weight
// matrices (rows x K) are exactly B^T. Lane n=lane&15 holds
// K values k0 + (lane>>4)*8 .. +7 and the same +16.
__device__ __forceinline__ v16h load_b_frag(const f16* __restrict__ tile,
                                            int ld, int k0, int lane) {
  const int n  = lane & 15;
  const int kb = k0 + ((lane >> 4) << 3);
  const f16* p = tile + (size_t)n * ld + kb;
  h8 lo = *(const h8*)(p);
  h8 hi = *(const h8*)(p + 16);
  v16h f;
#pragma unroll
  for (int i = 0; i < 8; ++i) { f[i] = lo[i]; f[i + 8] = hi[i]; }
  return f;
}

__device__ __forceinline__ v8f wmma16x16x32(v16h a, v16h b, v8f c) {
  return __builtin_amdgcn_wmma_f32_16x16x32_f16(false, a, false, b,
                                                (short)0, c, false, false);
}

__device__ __forceinline__ float sigm(float x) {
  return 1.0f / (1.0f + __expf(-x));
}

// ---------------------------------------------------------------------------
// Fused encoder LSTM step, both directions in one launch (blockIdx.y = dir).
// dir0 processes t=s (forward scan), dir1 processes t=L-1-s (reverse scan).
// One wave: 2 row-tiles x {i,f,g,o} gate-tiles, K-loop over Kin then H.
// Grid: (H/32, 2) blocks x 256 threads  => 4*(H/16) waves per direction.
// ---------------------------------------------------------------------------
__global__ void __launch_bounds__(256) lstm_step2(
    const f16* __restrict__ xseq,   // (L, 128, Kin)
    const f16* __restrict__ h_cur,  // (2, 128, H) ping
    f16* __restrict__ h_nxt,        // (2, 128, H) pong
    const f16* __restrict__ wih,    // (2, 4H, Kin)
    const f16* __restrict__ whh,    // (2, 4H, H)
    const float* __restrict__ bias, // (2, 4H)
    float* __restrict__ c_state,    // (2, 128, H) in/out (in-place safe)
    f16* __restrict__ outseq,       // (L, 128, 2H), dir writes its half
    int s, int Kin, int H)
{
  const int dir = blockIdx.y;
  const int t   = dir ? (L_SEQ - 1 - s) : s;
  const f16*   x_t    = xseq + (size_t)t * N_B * Kin;
  const f16*   wih_d  = wih + (size_t)dir * 4 * H * Kin;
  const f16*   whh_d  = whh + (size_t)dir * 4 * H * H;
  const float* bias_d = bias + (size_t)dir * 4 * H;
  float*       c_d    = c_state + (size_t)dir * N_B * H;
  const f16*   h_prev = h_cur + (size_t)dir * N_B * H;
  f16*         h_next = h_nxt + (size_t)dir * N_B * H;
  f16*         h_seq  = outseq + (size_t)t * N_B * (2 * HE_D) + (size_t)dir * HE_D;

  const int lane  = threadIdx.x & 31;
  const int wave  = (blockIdx.x * blockDim.x + threadIdx.x) >> 5;
  const int numJT = H >> 4;
  const int rtp   = wave / numJT;        // 0..3 : pair of row tiles
  const int jt    = wave - rtp * numJT;
  const int row0  = rtp * 32;

  v8f a0l = {}, a1l = {}, a2l = {}, a3l = {};
  v8f a0h = {}, a1h = {}, a2h = {}, a3h = {};

  // input contribution
  {
    const f16* ar0 = x_t + (size_t)row0 * Kin;
    const f16* ar1 = ar0 + (size_t)16 * Kin;
    const f16* w0 = wih_d + (size_t)(jt * 16) * Kin;
    const f16* w1 = wih_d + (size_t)(H + jt * 16) * Kin;
    const f16* w2 = wih_d + (size_t)(2 * H + jt * 16) * Kin;
    const f16* w3 = wih_d + (size_t)(3 * H + jt * 16) * Kin;
    for (int k0 = 0; k0 < Kin; k0 += 32) {
      v16h al = load_a_frag(ar0, Kin, k0, lane);
      v16h ah = load_a_frag(ar1, Kin, k0, lane);
      v16h b;
      b = load_b_frag(w0, Kin, k0, lane);
      a0l = wmma16x16x32(al, b, a0l); a0h = wmma16x16x32(ah, b, a0h);
      b = load_b_frag(w1, Kin, k0, lane);
      a1l = wmma16x16x32(al, b, a1l); a1h = wmma16x16x32(ah, b, a1h);
      b = load_b_frag(w2, Kin, k0, lane);
      a2l = wmma16x16x32(al, b, a2l); a2h = wmma16x16x32(ah, b, a2h);
      b = load_b_frag(w3, Kin, k0, lane);
      a3l = wmma16x16x32(al, b, a3l); a3h = wmma16x16x32(ah, b, a3h);
    }
  }
  // recurrent contribution
  {
    const f16* ar0 = h_prev + (size_t)row0 * H;
    const f16* ar1 = ar0 + (size_t)16 * H;
    const f16* u0 = whh_d + (size_t)(jt * 16) * H;
    const f16* u1 = whh_d + (size_t)(H + jt * 16) * H;
    const f16* u2 = whh_d + (size_t)(2 * H + jt * 16) * H;
    const f16* u3 = whh_d + (size_t)(3 * H + jt * 16) * H;
    for (int k0 = 0; k0 < H; k0 += 32) {
      v16h al = load_a_frag(ar0, H, k0, lane);
      v16h ah = load_a_frag(ar1, H, k0, lane);
      v16h b;
      b = load_b_frag(u0, H, k0, lane);
      a0l = wmma16x16x32(al, b, a0l); a0h = wmma16x16x32(ah, b, a0h);
      b = load_b_frag(u1, H, k0, lane);
      a1l = wmma16x16x32(al, b, a1l); a1h = wmma16x16x32(ah, b, a1h);
      b = load_b_frag(u2, H, k0, lane);
      a2l = wmma16x16x32(al, b, a2l); a2h = wmma16x16x32(ah, b, a2h);
      b = load_b_frag(u3, H, k0, lane);
      a3l = wmma16x16x32(al, b, a3l); a3h = wmma16x16x32(ah, b, a3h);
    }
  }

  const int hl = lane >> 4;
  const int j  = jt * 16 + (lane & 15);
  const float bi = bias_d[j], bf = bias_d[H + j];
  const float bg = bias_d[2 * H + j], bo = bias_d[3 * H + j];

  auto epilog = [&](v8f A0, v8f A1, v8f A2, v8f A3, int rbase) {
#pragma unroll
    for (int v = 0; v < 8; ++v) {
      const int n = rbase + v + 8 * hl;          // D row = v + 8*(lane>>4)
      const size_t idx = (size_t)n * H + j;
      const float gi = A0[v] + bi, gf = A1[v] + bf;
      const float gg = A2[v] + bg, go = A3[v] + bo;
      const float c  = c_d[idx];
      const float cn = sigm(gf) * c + sigm(gi) * tanhf(gg);
      const float hv = sigm(go) * tanhf(cn);
      c_d[idx]    = cn;
      h_next[idx] = (f16)hv;
      h_seq[(size_t)n * (2 * HE_D) + j] = (f16)hv;
    }
  };
  epilog(a0l, a1l, a2l, a3l, row0);
  epilog(a0h, a1h, a2h, a3h, row0 + 16);
}

// ---------------------------------------------------------------------------
// Decoder LSTM step. One-hot input GEMM collapses to a row-gather of dec_wih;
// recurrent GEMM on h_in = h_prev + enc_t (prepared by dec_pre).
// Grid: 32 blocks x 256 threads (4 row-pairs x 64 hidden tiles).
// ---------------------------------------------------------------------------
__global__ void __launch_bounds__(256) dec_step(
    const f16* __restrict__ h_in,     // (128, HD) f16
    const f16* __restrict__ whh,      // (4HD, HD) f16
    const float* __restrict__ wih,    // (4HD, T) fp32 (original)
    const float* __restrict__ bias,   // (4HD)
    const int* __restrict__ tag_prev, // (128)
    int has_prev,
    float* __restrict__ c_state,      // (128, HD) in/out
    float* __restrict__ h_out,        // (128, HD) fp32
    f16* __restrict__ h_out16)        // (128, HD) f16
{
  const int H = HD_D;
  const int lane  = threadIdx.x & 31;
  const int wave  = (blockIdx.x * blockDim.x + threadIdx.x) >> 5;
  const int numJT = H >> 4;                       // 64
  const int rtp   = wave / numJT;                 // 0..3
  const int jt    = wave - rtp * numJT;
  const int row0  = rtp * 32;

  v8f a0l = {}, a1l = {}, a2l = {}, a3l = {};
  v8f a0h = {}, a1h = {}, a2h = {}, a3h = {};
  const f16* ar0 = h_in + (size_t)row0 * H;
  const f16* ar1 = ar0 + (size_t)16 * H;
  const f16* u0 = whh + (size_t)(jt * 16) * H;
  const f16* u1 = whh + (size_t)(H + jt * 16) * H;
  const f16* u2 = whh + (size_t)(2 * H + jt * 16) * H;
  const f16* u3 = whh + (size_t)(3 * H + jt * 16) * H;
  for (int k0 = 0; k0 < H; k0 += 32) {
    v16h al = load_a_frag(ar0, H, k0, lane);
    v16h ah = load_a_frag(ar1, H, k0, lane);
    v16h b;
    b = load_b_frag(u0, H, k0, lane);
    a0l = wmma16x16x32(al, b, a0l); a0h = wmma16x16x32(ah, b, a0h);
    b = load_b_frag(u1, H, k0, lane);
    a1l = wmma16x16x32(al, b, a1l); a1h = wmma16x16x32(ah, b, a1h);
    b = load_b_frag(u2, H, k0, lane);
    a2l = wmma16x16x32(al, b, a2l); a2h = wmma16x16x32(ah, b, a2h);
    b = load_b_frag(u3, H, k0, lane);
    a3l = wmma16x16x32(al, b, a3l); a3h = wmma16x16x32(ah, b, a3h);
  }

  const int hl = lane >> 4;
  const int j  = jt * 16 + (lane & 15);
  const float bi = bias[j], bf = bias[H + j];
  const float bg = bias[2 * H + j], bo = bias[3 * H + j];

  auto epilog = [&](v8f A0, v8f A1, v8f A2, v8f A3, int rbase) {
#pragma unroll
    for (int v = 0; v < 8; ++v) {
      const int n = rbase + v + 8 * hl;
      const size_t idx = (size_t)n * H + j;
      float gi = A0[v] + bi, gf = A1[v] + bf;
      float gg = A2[v] + bg, go = A3[v] + bo;
      if (has_prev) {
        const int tg = tag_prev[n];               // one-hot @ Wih^T == gather
        gi += wih[(size_t)j * T_TAG + tg];
        gf += wih[(size_t)(H + j) * T_TAG + tg];
        gg += wih[(size_t)(2 * H + j) * T_TAG + tg];
        go += wih[(size_t)(3 * H + j) * T_TAG + tg];
      }
      const float c  = c_state[idx];
      const float cn = sigm(gf) * c + sigm(gi) * tanhf(gg);
      const float hv = sigm(go) * tanhf(cn);
      c_state[idx] = cn;
      h_out[idx]   = hv;
      h_out16[idx] = (f16)hv;
    }
  };
  epilog(a0l, a1l, a2l, a3l, row0);
  epilog(a0h, a1h, a2h, a3h, row0 + 16);
}

// logits = h @ out_w^T + out_b  (128 x 128, K=1024).  8 blocks x 256 threads.
__global__ void __launch_bounds__(256) logits_gemm(
    const f16* __restrict__ h16,  // (128, HD)
    const f16* __restrict__ wo,   // (T, HD)  == B^T row-major
    const float* __restrict__ ob, // (T)
    float* __restrict__ logits)   // (128, T)
{
  const int lane = threadIdx.x & 31;
  const int wave = (blockIdx.x * blockDim.x + threadIdx.x) >> 5; // 0..63
  const int rt = wave >> 3;
  const int ct = wave & 7;
  v8f acc = {};
  const f16* arow = h16 + (size_t)(rt * 16) * HD_D;
  const f16* brow = wo + (size_t)(ct * 16) * HD_D;
  for (int k0 = 0; k0 < HD_D; k0 += 32) {
    v16h a = load_a_frag(arow, HD_D, k0, lane);
    v16h b = load_b_frag(brow, HD_D, k0, lane);
    acc = wmma16x16x32(a, b, acc);
  }
  const int hl  = lane >> 4;
  const int col = ct * 16 + (lane & 15);
  const float bb = ob[col];
#pragma unroll
  for (int v = 0; v < 8; ++v) {
    const int n = rt * 16 + v + 8 * hl;
    logits[n * T_TAG + col] = acc[v] + bb;
  }
}

// Per-row softmax + prob write + greedy argmax feedback + loss accumulation.
// Single block => deterministic fp accumulation order across the 160 steps.
__global__ void __launch_bounds__(128) softmax_loss(
    const float* __restrict__ logits, const int* __restrict__ tag_ids,
    int t, float* __restrict__ prob_out, float* __restrict__ loss_ptr,
    int* __restrict__ tag_next)
{
  const int n = threadIdx.x;
  const float* row = logits + n * T_TAG;
  float mx = row[0]; int am = 0;
  for (int k = 1; k < T_TAG; ++k) { float v = row[k]; if (v > mx) { mx = v; am = k; } }
  float se = 0.f;
  for (int k = 0; k < T_TAG; ++k) se += __expf(row[k] - mx);
  const float lz = mx + __logf(se);
  float* op = prob_out + (size_t)n * L_SEQ * T_TAG + (size_t)t * T_TAG;
  for (int k = 0; k < T_TAG; ++k) op[k] = __expf(row[k] - lz);
  const int tg = tag_ids[n * L_SEQ + t];
  const float myl = -(row[tg] - lz) * (1.0f / (float)N_B);
  __shared__ float red[128];
  red[n] = myl; __syncthreads();
  for (int s = 64; s > 0; s >>= 1) { if (n < s) red[n] += red[n + s]; __syncthreads(); }
  if (n == 0) atomicAdd(loss_ptr, red[0]);
  tag_next[n] = am;
}

// ------------------------------- small utils -------------------------------
__global__ void f32_to_f16_k(const float* __restrict__ s, f16* __restrict__ d, int n) {
  int i = blockIdx.x * blockDim.x + threadIdx.x;
  if (i < n) d[i] = (f16)s[i];
}
__global__ void embed_gather(const int* __restrict__ ids, const float* __restrict__ emb,
                             f16* __restrict__ x) {
  size_t i = (size_t)blockIdx.x * blockDim.x + threadIdx.x;
  if (i >= (size_t)L_SEQ * N_B * E_DIM) return;
  const int e = (int)(i % E_DIM);
  const int n = (int)((i / E_DIM) % N_B);
  const int t = (int)(i / ((size_t)E_DIM * N_B));
  const int vid = ids[n * L_SEQ + t];
  x[i] = (f16)emb[(size_t)vid * E_DIM + e];
}
__global__ void zero_state(f16* h, float* c, int n) {
  int i = blockIdx.x * blockDim.x + threadIdx.x;
  if (i < n) { h[i] = (f16)0.f; c[i] = 0.f; }
}
// h0=0; c0 = concat([enc[0,:,HE:], enc[0,:,HE:]]); loss=0
__global__ void dec_init(const f16* __restrict__ enc0, float* h, float* c,
                         float* loss) {
  int i = blockIdx.x * blockDim.x + threadIdx.x;
  if (i < N_B * HD_D) {
    const int n = i / HD_D, j = i % HD_D;
    h[i] = 0.f;
    c[i] = (float)enc0[(size_t)n * HD_D + HE_D + (j & (HE_D - 1))];
  }
  if (i == 0) *loss = 0.f;
}
__global__ void dec_pre(const float* __restrict__ h, const f16* __restrict__ enc_t,
                        f16* __restrict__ hin) {
  int i = blockIdx.x * blockDim.x + threadIdx.x;
  if (i < N_B * HD_D) hin[i] = (f16)(h[i] + (float)enc_t[i]);
}

// ---------------------------------------------------------------------------
extern "C" void kernel_launch(void* const* d_in, const int* in_sizes, int n_in,
                              void* d_out, int out_size, void* d_ws, size_t ws_size,
                              hipStream_t stream) {
  (void)in_sizes; (void)n_in; (void)out_size; (void)ws_size;
  const int*   input_ids = (const int*)d_in[0];
  const int*   tag_ids   = (const int*)d_in[1];
  const float* embed     = (const float*)d_in[2];
  const float* enc0_wih  = (const float*)d_in[3];
  const float* enc0_whh  = (const float*)d_in[4];
  const float* enc0_b    = (const float*)d_in[5];
  const float* enc1_wih  = (const float*)d_in[6];
  const float* enc1_whh  = (const float*)d_in[7];
  const float* enc1_b    = (const float*)d_in[8];
  const float* dec_wih   = (const float*)d_in[9];
  const float* dec_whh   = (const float*)d_in[10];
  const float* dec_b     = (const float*)d_in[11];
  const float* out_w     = (const float*)d_in[12];
  const float* out_b     = (const float*)d_in[13];

  char* p = (char*)d_ws;
  auto alloc = [&](size_t elems, size_t esz) {
    void* r = (void*)p; p += (elems * esz + 255) & ~(size_t)255; return r;
  };
  // f16 weights (~31 MB total -> L2-resident)
  f16* w0_wih = (f16*)alloc((size_t)2 * 4 * HE_D * E_DIM, 2);
  f16* w0_whh = (f16*)alloc((size_t)2 * 4 * HE_D * HE_D, 2);
  f16* w1_wih = (f16*)alloc((size_t)2 * 4 * HE_D * 2 * HE_D, 2);
  f16* w1_whh = (f16*)alloc((size_t)2 * 4 * HE_D * HE_D, 2);
  f16* wd_whh = (f16*)alloc((size_t)4 * HD_D * HD_D, 2);
  f16* wo     = (f16*)alloc((size_t)T_TAG * HD_D, 2);
  // sequences (f16)
  f16* x16 = (f16*)alloc((size_t)L_SEQ * N_B * E_DIM, 2);    // (L,N,E)
  f16* h1  = (f16*)alloc((size_t)L_SEQ * N_B * 2 * HE_D, 2); // (L,N,1024)
  f16* enc = (f16*)alloc((size_t)L_SEQ * N_B * 2 * HE_D, 2); // (L,N,1024)
  // states (enc states hold both directions: (2, N, HE))
  f16*   hA     = (f16*)alloc((size_t)2 * N_B * HE_D, 2);
  f16*   hB     = (f16*)alloc((size_t)2 * N_B * HE_D, 2);
  float* cE     = (float*)alloc((size_t)2 * N_B * HE_D, 4);
  float* hdec   = (float*)alloc((size_t)N_B * HD_D, 4);
  float* cdec   = (float*)alloc((size_t)N_B * HD_D, 4);
  f16*   hin    = (f16*)alloc((size_t)N_B * HD_D, 2);
  f16*   hdec16 = (f16*)alloc((size_t)N_B * HD_D, 2);
  float* logits = (float*)alloc((size_t)N_B * T_TAG, 4);
  int*   tagbuf = (int*)alloc(N_B, 4);

  float* prob_out = (float*)d_out;
  float* loss_ptr = (float*)d_out + (size_t)N_B * L_SEQ * T_TAG;

  auto cvt = [&](const float* s, f16* d, int n) {
    f32_to_f16_k<<<(n + 255) / 256, 256, 0, stream>>>(s, d, n);
  };
  cvt(enc0_wih, w0_wih, 2 * 4 * HE_D * E_DIM);
  cvt(enc0_whh, w0_whh, 2 * 4 * HE_D * HE_D);
  cvt(enc1_wih, w1_wih, 2 * 4 * HE_D * 2 * HE_D);
  cvt(enc1_whh, w1_whh, 2 * 4 * HE_D * HE_D);
  cvt(dec_whh, wd_whh, 4 * HD_D * HD_D);
  cvt(out_w, wo, T_TAG * HD_D);

  {
    const size_t nx = (size_t)L_SEQ * N_B * E_DIM;
    embed_gather<<<(int)((nx + 255) / 256), 256, 0, stream>>>(input_ids, embed, x16);
  }

  // one bidirectional layer: both directions per launch, sequential over L
  auto run_layer = [&](const f16* xseq, int Kin, const f16* wih, const f16* whh,
                       const float* bias, f16* outseq) {
    zero_state<<<(2 * N_B * HE_D + 255) / 256, 256, 0, stream>>>(hA, cE,
                                                                 2 * N_B * HE_D);
    f16 *cur = hA, *nxt = hB;
    const dim3 grid(HE_D / 32, 2);   // (numJT/2 blocks, 2 directions)
    for (int s = 0; s < L_SEQ; ++s) {
      lstm_step2<<<grid, 256, 0, stream>>>(xseq, cur, nxt, wih, whh, bias, cE,
                                           outseq, s, Kin, HE_D);
      f16* tmp = cur; cur = nxt; nxt = tmp;
    }
  };
  run_layer(x16, E_DIM, w0_wih, w0_whh, enc0_b, h1);       // encoder layer 0
  run_layer(h1, 2 * HE_D, w1_wih, w1_whh, enc1_b, enc);    // encoder layer 1

  // decoder
  dec_init<<<(N_B * HD_D + 255) / 256, 256, 0, stream>>>(enc, hdec, cdec, loss_ptr);
  for (int t = 0; t < L_SEQ; ++t) {
    dec_pre<<<(N_B * HD_D + 255) / 256, 256, 0, stream>>>(
        hdec, enc + (size_t)t * N_B * HD_D, hin);
    dec_step<<<HD_D / 32, 256, 0, stream>>>(hin, wd_whh, dec_wih, dec_b, tagbuf,
                                            t > 0 ? 1 : 0, cdec, hdec, hdec16);
    logits_gemm<<<8, 256, 0, stream>>>(hdec16, wo, out_b, logits);
    softmax_loss<<<1, 128, 0, stream>>>(logits, tag_ids, t, prob_out, loss_ptr,
                                        tagbuf);
  }
}
